// AutoformerAttention_49228915146930
// MI455X (gfx1250) — compile-verified, matrix operand order checked by reading, not run on hardware
//
#include <hip/hip_runtime.h>
#include <hip/hip_bf16.h>

#define TB 4
#define TT 4096
#define TE 512
#define THN 8
#define TDD 64
#define TOPK 24
#define NROWS (TB*TT)          /* 16384 */
#define CHG 8                  /* channels per FFT group */
#define NG (TE/CHG)            /* 64 groups per batch */

typedef unsigned short u16t;
typedef __attribute__((ext_vector_type(16))) __bf16 v16bf;
typedef __attribute__((ext_vector_type(8)))  float  v8f;
typedef unsigned int v4u __attribute__((ext_vector_type(4)));
typedef int          v4i __attribute__((ext_vector_type(4)));
typedef int          v8i __attribute__((ext_vector_type(8)));

struct U128 { unsigned int a, b, c, d; };

#if defined(__has_builtin)
#  if __has_builtin(__builtin_amdgcn_tensor_load_to_lds) && __has_builtin(__builtin_amdgcn_s_wait_tensorcnt)
#    define HAVE_TDM 1
#  endif
#endif
#ifndef HAVE_TDM
#  define HAVE_TDM 0
#endif

__device__ inline u16t f2bf(float f) {
  union { float f; unsigned u; } v; v.f = f;
  unsigned r = v.u + 0x7FFFu + ((v.u >> 16) & 1u);
  return (u16t)(r >> 16);
}

__device__ inline float2 cmulf(float2 a, float2 b) {
  float2 r; r.x = a.x*b.x - a.y*b.y; r.y = a.x*b.y + a.y*b.x; return r;
}

// ---------------- fp32 -> bf16 conversion ----------------
__global__ void k_cvt_bf16(const float* __restrict__ src, u16t* __restrict__ dst, int n) {
  int i = blockIdx.x * blockDim.x + threadIdx.x;
  if (i < n) dst[i] = f2bf(src[i]);
}

// ---------------- WMMA fragment load from global (two b128 loads) ----------------
__device__ inline v16bf load_frag(const u16t* __restrict__ mat, int row0, int lane, int kt, int ld) {
  int lr = lane & 15, hi = lane >> 4;
  const u16t* p = mat + (size_t)(row0 + lr) * ld + kt + hi * 8;
  union { v16bf v; U128 q[2]; } u;
  u.q[0] = *(const U128*)(p);
  u.q[1] = *(const U128*)(p + 16);
  return u.v;
}

// ---------------- WMMA fragment load from LDS slab (two ds_load_b128) ----------------
__device__ inline v16bf load_frag_lds(const u16t* ptr, int lane, int kt) {
  int lr = lane & 15, hi = lane >> 4;
  const u16t* p = ptr + lr * TE + kt + hi * 8;
  union { v16bf v; U128 q[2]; } u;
  u.q[0] = *(const U128*)(p);
  u.q[1] = *(const U128*)(p + 16);
  return u.v;
}

#if HAVE_TDM
// TDM: DMA a contiguous 16-row x 512-col bf16 slab (row-major) into LDS.
// D# per CDNA5 ISA 8.3-8.6: group0 = {count/type, lds_addr, global_addr},
// group1 = {data_size, tensor_dim0/1, tile_dim0/1, stride0}. 2D tensor.
__device__ inline void tdm_load_w16x512(const u16t* gptr, unsigned lds_off) {
  unsigned long long ga = (unsigned long long)(uintptr_t)gptr;
  v4u g0;
  g0[0] = 1u;                                              // count=1 (user descriptor)
  g0[1] = lds_off;                                         // lds_addr (bytes)
  g0[2] = (unsigned)(ga & 0xFFFFFFFFu);                    // global_addr[31:0]
  g0[3] = (unsigned)((ga >> 32) & 0x01FFFFFFu) | 0x80000000u; // addr[56:32] | type=2
  v8i g1;
  g1[0] = 0x00010000;                                      // data_size=1 (2 bytes/elem)
  g1[1] = (int)(512u << 16);                               // tensor_dim0 = 512
  g1[2] = (int)(512u << 16);                               // tensor_dim1 = 512
  g1[3] = (int)(512u << 16);                               // tile_dim0 = 512
  g1[4] = 16;                                              // tile_dim1 = 16
  g1[5] = 512;                                             // tensor_dim0_stride = 512
  g1[6] = 0;
  g1[7] = 0;
  v4i z4 = {0, 0, 0, 0};
#if __clang_major__ >= 23
  v8i z8 = {0, 0, 0, 0, 0, 0, 0, 0};
  __builtin_amdgcn_tensor_load_to_lds(g0, g1, z4, z4, z8, 0);
#else
  __builtin_amdgcn_tensor_load_to_lds(g0, g1, z4, z4, 0);
#endif
}
#endif

// ---------------- fused QKV projection: TDM-staged weights in LDS ----------------
__global__ __launch_bounds__(256) void k_qkv(
    const u16t* __restrict__ xbf,
    const u16t* __restrict__ wqm, const u16t* __restrict__ wkm, const u16t* __restrict__ wvm,
    const float* __restrict__ bq, const float* __restrict__ bk, const float* __restrict__ bv,
    float* __restrict__ qT, float* __restrict__ kT, float* __restrict__ vout)
{
  __shared__ u16t wlds[3 * 16 * TE];   // 48 KB: Wq/Wk/Wv 16-row slabs for this n-block
  int lane = threadIdx.x & 31;
  int wid  = threadIdx.x >> 5;
  int m0 = (blockIdx.x * 8 + wid) * 16;
  int n0 = blockIdx.y * 16;

#if HAVE_TDM
  if (threadIdx.x == 0) {
    unsigned base = (unsigned)(uintptr_t)&wlds[0];
    tdm_load_w16x512(wqm + (size_t)n0 * TE, base);
    tdm_load_w16x512(wkm + (size_t)n0 * TE, base + 16 * TE * 2);
    tdm_load_w16x512(wvm + (size_t)n0 * TE, base + 2 * 16 * TE * 2);
    __builtin_amdgcn_s_wait_tensorcnt(0);
  }
#else
  {
    const u16t* s0 = wqm + (size_t)n0 * TE;
    const u16t* s1 = wkm + (size_t)n0 * TE;
    const u16t* s2 = wvm + (size_t)n0 * TE;
    for (int c = threadIdx.x; c < 3 * 1024; c += 256) {
      int m = c >> 10, o = (c & 1023) * 8;
      const u16t* s = (m == 0) ? s0 : (m == 1) ? s1 : s2;
      *(U128*)(&wlds[m * 16 * TE + o]) = *(const U128*)(s + o);
    }
  }
#endif
  __syncthreads();

  v8f aq = {}, ak = {}, av = {};
  for (int kt = 0; kt < TE; kt += 32) {
    if (kt + 32 < TE)
      __builtin_prefetch(xbf + (size_t)m0 * TE + kt + 32, 0, 1);
    v16bf a = load_frag(xbf, m0, lane, kt, TE);
    v16bf fb;
    fb = load_frag_lds(&wlds[0],           lane, kt);
    aq = __builtin_amdgcn_wmma_f32_16x16x32_bf16(false, a, false, fb, (short)0, aq, false, false);
    fb = load_frag_lds(&wlds[16 * TE],     lane, kt);
    ak = __builtin_amdgcn_wmma_f32_16x16x32_bf16(false, a, false, fb, (short)0, ak, false, false);
    fb = load_frag_lds(&wlds[2 * 16 * TE], lane, kt);
    av = __builtin_amdgcn_wmma_f32_16x16x32_bf16(false, a, false, fb, (short)0, av, false, false);
  }
  int lr = lane & 15, hi = lane >> 4;
  int n = n0 + lr;
  float qb = bq[n], kb = bk[n], vb = bv[n];
#pragma unroll
  for (int r = 0; r < 8; ++r) {
    int m = m0 + r + 8 * hi;
    qT[(size_t)n * NROWS + m] = aq[r] + qb;
    kT[(size_t)n * NROWS + m] = ak[r] + kb;
    vout[(size_t)m * TE + n]  = av[r] + vb;
  }
}

// ---------------- in-LDS radix-2 DIF FFT (4096 pt, natural in -> bit-reversed out) ----------------
__device__ inline void fft_dif_4096(float2* z, int tid) {
  for (int lm = 11; lm >= 0; --lm) {
    int m = 1 << lm;
    for (int idx = tid; idx < TT / 2; idx += 256) {
      int jj  = idx & (m - 1);
      int blk = idx >> lm;
      int i0  = (blk << (lm + 1)) + jj;
      float2 a = z[i0], bb = z[i0 + m];
      float2 s; s.x = a.x + bb.x; s.y = a.y + bb.y;
      float2 d; d.x = a.x - bb.x; d.y = a.y - bb.y;
      float ang = -3.14159265358979323846f * (float)jj / (float)m;
      float sn, cs; __sincosf(ang, &sn, &cs);
      float2 w; w.x = cs; w.y = sn;
      z[i0] = s;
      z[i0 + m] = cmulf(d, w);
    }
    __syncthreads();
  }
}

// ---------------- FFT cross-spectrum: Z=FFT(q+ik); accumulate Qf*conj(Kf) per group ----------------
__global__ __launch_bounds__(256) void k_fft_corr(const float* __restrict__ qT,
                                                  const float* __restrict__ kT,
                                                  float2* __restrict__ part)
{
  __shared__ float2 z[TT];          // 32 KB
  int tid = threadIdx.x;
  int g = blockIdx.x, b = blockIdx.y;
  float2 acc[16];
#pragma unroll
  for (int j = 0; j < 16; ++j) { acc[j].x = 0.f; acc[j].y = 0.f; }

  for (int c = 0; c < CHG; ++c) {
    int e = g * CHG + c;
    const float* qp = qT + (size_t)e * NROWS + b * TT;
    const float* kp = kT + (size_t)e * NROWS + b * TT;
    __syncthreads();
    for (int j = tid; j < TT; j += 256) { float2 t; t.x = qp[j]; t.y = kp[j]; z[j] = t; }
    __syncthreads();
    fft_dif_4096(z, tid);
    // natural-frequency-order extraction from bit-reversed storage
#pragma unroll
    for (int j = 0; j < 16; ++j) {
      int f   = j * 256 + tid;
      int fr  = (int)(__brev((unsigned)f) >> 20);
      int nfr = (int)(__brev((unsigned)((TT - f) & (TT - 1))) >> 20);
      float2 Zf = z[fr], Zn = z[nfr];
      float2 Q;  Q.x = 0.5f * (Zf.x + Zn.x); Q.y = 0.5f * (Zf.y - Zn.y);
      float2 W;  W.x = Zf.x - Zn.x;          W.y = Zf.y + Zn.y;
      float2 Kc; Kc.x = 0.5f * W.y;          Kc.y = 0.5f * W.x;  // conj(K)
      float2 p = cmulf(Q, Kc);
      acc[j].x += p.x; acc[j].y += p.y;
    }
  }
  float2* out = part + ((size_t)(b * NG + g)) * TT;
#pragma unroll
  for (int j = 0; j < 16; ++j) out[j * 256 + tid] = acc[j];
}

// ---------------- reduce partial spectra + inverse FFT -> ac_mean[b][t] ----------------
__global__ __launch_bounds__(256) void k_reduce_ifft(const float2* __restrict__ part,
                                                     float* __restrict__ acm)
{
  __shared__ float2 z[TT];
  int tid = threadIdx.x, b = blockIdx.x;
#pragma unroll
  for (int j = 0; j < 16; ++j) {
    int f = j * 256 + tid;
    float2 s; s.x = 0.f; s.y = 0.f;
    for (int g = 0; g < NG; ++g) {
      float2 p = part[((size_t)(b * NG + g)) * TT + f];
      s.x += p.x; s.y += p.y;
    }
    z[f].x = s.x; z[f].y = -s.y;   // conj -> inverse via forward transform
  }
  __syncthreads();
  fft_dif_4096(z, tid);
  const float scale = 1.0f / ((float)TT * (float)(THN * TDD));
#pragma unroll
  for (int j = 0; j < 16; ++j) {
    int t  = j * 256 + tid;
    int tr = (int)(__brev((unsigned)t) >> 20);
    acm[b * TT + t] = z[tr].x * scale;
  }
}

// ---------------- per-batch top-24 + softmax ----------------
__global__ __launch_bounds__(256) void k_topk(const float* __restrict__ acm,
                                              float* __restrict__ wout, int* __restrict__ dout)
{
  __shared__ float sv[TT];
  __shared__ float rv[256];
  __shared__ int   ri[256];
  __shared__ float tv[TOPK];
  __shared__ int   ti[TOPK];
  int tid = threadIdx.x, b = blockIdx.x;
  for (int j = tid; j < TT; j += 256) sv[j] = acm[b * TT + j];
  __syncthreads();
  for (int it = 0; it < TOPK; ++it) {
    float best = -3.0e38f; int bi = 0;
    for (int j = tid; j < TT; j += 256) { float x = sv[j]; if (x > best) { best = x; bi = j; } }
    rv[tid] = best; ri[tid] = bi;
    __syncthreads();
    for (int s = 128; s > 0; s >>= 1) {
      if (tid < s && rv[tid + s] > rv[tid]) { rv[tid] = rv[tid + s]; ri[tid] = ri[tid + s]; }
      __syncthreads();
    }
    if (tid == 0) { tv[it] = rv[0]; ti[it] = ri[0]; sv[ri[0]] = -3.0e38f; }
    __syncthreads();
  }
  if (tid == 0) {
    float mx = tv[0];
    for (int i = 1; i < TOPK; ++i) mx = fmaxf(mx, tv[i]);
    float ex[TOPK], sum = 0.f;
    for (int i = 0; i < TOPK; ++i) { ex[i] = __expf(tv[i] - mx); sum += ex[i]; }
    float inv = 1.f / sum;
    for (int i = 0; i < TOPK; ++i) { wout[b * TOPK + i] = ex[i] * inv; dout[b * TOPK + i] = ti[i]; }
  }
}

// ---------------- weighted circular-shift aggregation of v, fused bf16 convert ----------------
__global__ __launch_bounds__(256) void k_agg(const float* __restrict__ v,
                                             const float* __restrict__ w,
                                             const int* __restrict__ dly,
                                             u16t* __restrict__ aggbf)
{
  long long gid = (long long)blockIdx.x * blockDim.x + threadIdx.x;
  if (gid >= (long long)NROWS * TE) return;
  int e = (int)(gid % TE);
  long long row = gid / TE;
  int b = (int)(row / TT);
  int t = (int)(row % TT);
  int h = e >> 6;
  int jb = (b * THN + h) % TB;   // torch .repeat tiling quirk
  float acc = 0.f;
#pragma unroll
  for (int i = 0; i < TOPK; ++i) {
    int d  = dly[jb * TOPK + i];
    int tt = (t + d) & (TT - 1);
    acc += w[jb * TOPK + i] * v[((size_t)b * TT + tt) * TE + e];
  }
  aggbf[gid] = f2bf(acc);
}

// ---------------- output projection GEMM (WMMA bf16, TDM-staged Wo) ----------------
__global__ __launch_bounds__(256) void k_out(const u16t* __restrict__ abf,
                                             const u16t* __restrict__ wom,
                                             const float* __restrict__ bo,
                                             float* __restrict__ out)
{
  __shared__ u16t wlds[16 * TE];   // 16 KB: Wo 16-row slab
  int lane = threadIdx.x & 31;
  int wid  = threadIdx.x >> 5;
  int m0 = (blockIdx.x * 8 + wid) * 16;
  int n0 = blockIdx.y * 16;

#if HAVE_TDM
  if (threadIdx.x == 0) {
    tdm_load_w16x512(wom + (size_t)n0 * TE, (unsigned)(uintptr_t)&wlds[0]);
    __builtin_amdgcn_s_wait_tensorcnt(0);
  }
#else
  {
    const u16t* s = wom + (size_t)n0 * TE;
    for (int c = threadIdx.x; c < 1024; c += 256) {
      int o = c * 8;
      *(U128*)(&wlds[o]) = *(const U128*)(s + o);
    }
  }
#endif
  __syncthreads();

  v8f acc = {};
  for (int kt = 0; kt < TE; kt += 32) {
    if (kt + 32 < TE)
      __builtin_prefetch(abf + (size_t)m0 * TE + kt + 32, 0, 1);
    v16bf a  = load_frag(abf, m0, lane, kt, TE);
    v16bf fb = load_frag_lds(&wlds[0], lane, kt);
    acc = __builtin_amdgcn_wmma_f32_16x16x32_bf16(false, a, false, fb, (short)0, acc, false, false);
  }
  int lr = lane & 15, hi = lane >> 4;
  int n = n0 + lr;
  float bb = bo[n];
#pragma unroll
  for (int r = 0; r < 8; ++r) {
    int m = m0 + r + 8 * hi;
    out[(size_t)m * TE + n] = acc[r] + bb;
  }
}

extern "C" void kernel_launch(void* const* d_in, const int* in_sizes, int n_in,
                              void* d_out, int out_size, void* d_ws, size_t ws_size,
                              hipStream_t stream)
{
  (void)in_sizes; (void)n_in; (void)out_size; (void)ws_size;
  const float* x  = (const float*)d_in[0];
  const float* Wq = (const float*)d_in[1];
  const float* bq = (const float*)d_in[2];
  const float* Wk = (const float*)d_in[3];
  const float* bk = (const float*)d_in[4];
  const float* Wv = (const float*)d_in[5];
  const float* bv = (const float*)d_in[6];
  const float* Wo = (const float*)d_in[7];
  const float* bo = (const float*)d_in[8];
  float* out = (float*)d_out;

  char* ws = (char*)d_ws;
  size_t off = 0;
  u16t* xbf = (u16t*)(ws + off); off += (size_t)NROWS * TE * 2;   // 16 MB (reused as agg bf16)
  u16t* wqb = (u16t*)(ws + off); off += (size_t)TE * TE * 2;
  u16t* wkb = (u16t*)(ws + off); off += (size_t)TE * TE * 2;
  u16t* wvb = (u16t*)(ws + off); off += (size_t)TE * TE * 2;
  u16t* wob = (u16t*)(ws + off); off += (size_t)TE * TE * 2;
  float*  qT   = (float*) (ws + off); off += (size_t)NROWS * TE * 4;   // 32 MB, [E][B*T]
  float*  kT   = (float*) (ws + off); off += (size_t)NROWS * TE * 4;   // 32 MB
  float*  vf   = (float*) (ws + off); off += (size_t)NROWS * TE * 4;   // 32 MB, [B*T][E]
  float2* part = (float2*)(ws + off); off += (size_t)TB * NG * TT * 8; // 8 MB
  float*  acm  = (float*) (ws + off); off += (size_t)TB * TT * 4;
  float*  wsm  = (float*) (ws + off); off += (size_t)TB * TOPK * 4;
  int*    dly  = (int*)   (ws + off); off += (size_t)TB * TOPK * 4;

  int n = NROWS * TE;
  k_cvt_bf16<<<(n + 255) / 256, 256, 0, stream>>>(x, xbf, n);
  n = TE * TE;
  k_cvt_bf16<<<(n + 255) / 256, 256, 0, stream>>>(Wq, wqb, n);
  k_cvt_bf16<<<(n + 255) / 256, 256, 0, stream>>>(Wk, wkb, n);
  k_cvt_bf16<<<(n + 255) / 256, 256, 0, stream>>>(Wv, wvb, n);
  k_cvt_bf16<<<(n + 255) / 256, 256, 0, stream>>>(Wo, wob, n);

  dim3 gg(NROWS / 128, TE / 16);
  k_qkv<<<gg, 256, 0, stream>>>(xbf, wqb, wkb, wvb, bq, bk, bv, qT, kT, vf);

  k_fft_corr<<<dim3(NG, TB), 256, 0, stream>>>(qT, kT, part);
  k_reduce_ifft<<<TB, 256, 0, stream>>>(part, acm);
  k_topk<<<TB, 256, 0, stream>>>(acm, wsm, dly);

  long long tot = (long long)NROWS * TE;
  k_agg<<<(unsigned)((tot + 255) / 256), 256, 0, stream>>>(vf, wsm, dly, xbf);

  k_out<<<gg, 256, 0, stream>>>(xbf, wob, bo, out);
}
